// GraphUNetModel_15796889715333
// MI455X (gfx1250) — compile-verified, matrix operand order checked by reading, not run on hardware
//
#include <hip/hip_runtime.h>
#include <hip/hip_bf16.h>

typedef __attribute__((ext_vector_type(16))) _Float16 v16h;
typedef __attribute__((ext_vector_type(8)))  _Float16 v8h;
typedef __attribute__((ext_vector_type(8)))  float    v8f;
typedef __attribute__((ext_vector_type(4)))  int      v4i;

#define BM 128
#define BN 128
#define BK 32
#define LDSS 40   // padded LDS stride in f16 elements (conflict-free frag reads)

// ---- gfx1250 async global->LDS (ASYNCcnt) with safe fallback ----------------
#if defined(__HIP_DEVICE_COMPILE__) && defined(__has_builtin)
#  if __has_builtin(__builtin_amdgcn_global_load_async_to_lds_b128)
#    define HAVE_ASYNC_LDS 1
#  endif
#endif
#ifndef HAVE_ASYNC_LDS
#  define HAVE_ASYNC_LDS 0
#endif

#if HAVE_ASYNC_LDS
typedef __attribute__((address_space(1))) v4i* gv4p;   // global int4*
typedef __attribute__((address_space(3))) v4i* lv4p;   // LDS int4*
__device__ __forceinline__ void async_cp16(const _Float16* g, _Float16* l)
{
    // prototype (from compiler diagnostic): (int4 AS1*, int4 AS3*, imm, imm)
    __builtin_amdgcn_global_load_async_to_lds_b128((gv4p)(v4i*)g, (lv4p)(v4i*)l, 0, 0);
}
#endif

__device__ __forceinline__ void wait_async0()
{
#if HAVE_ASYNC_LDS
    asm volatile("s_wait_asynccnt 0x0" ::: "memory");
#endif
}

// ---------------------------------------------------------------------------
// Tiled f16->f32 WMMA GEMM.  C(MxN) = A(MxK) @ B(KxN), all row-major.
// Block tile 128x128, 8 wave32 waves each owning 32x64 (2x4 WMMA frags),
// K step 32, double-buffered LDS (one barrier per K tile).
// MODE 0: plain store
// MODE 1: augment epilogue: zero the diagonal (C square, ldc == Ncols)
// MODE 2: GCN epilogue:   C = dinv[r]*(acc + 2*xws[r,c]) + bias[c], opt. leaky
// Requires M%128==0, Ncols%128==0, K%32==0 (true for all shapes here).
// ---------------------------------------------------------------------------
template<int MODE>
__global__ __launch_bounds__(256)
void gemm_f16(const _Float16* __restrict__ A, int lda,
              const _Float16* __restrict__ B, int ldb,
              float* __restrict__ C, int ldc,
              int M, int Ncols, int K,
              const float* __restrict__ dinv,
              const float* __restrict__ xws,
              const float* __restrict__ bias,
              int leaky)
{
    __shared__ __align__(16) _Float16 As[2][BM * LDSS];
    __shared__ __align__(16) _Float16 Bs[2][BN * LDSS];   // transposed: [n][k]

    const int tid   = threadIdx.x;
    const int lane  = tid & 31;
    const int wid   = tid >> 5;
    const int wm    = wid & 3;           // 0..3  (M direction, 32 rows each)
    const int wn    = wid >> 2;          // 0..1  (N direction, 64 cols each)
    const int l16   = lane & 15;
    const int lhalf = lane >> 4;

    const int rowBlock = blockIdx.y * BM;
    const int colBlock = blockIdx.x * BN;

    // A staging: thread -> (row, 16-elem k segment); two b128 per thread
    const int ar = tid >> 1;             // 0..127
    const int ak = (tid & 1) << 4;       // 0 / 16
    // B staging: thread -> (k pair, 8-col segment); two b128 per thread
    const int bkp = (tid >> 4) << 1;     // 0,2,..,30
    const int bn  = (tid & 15) << 3;     // 0..120

    v8f acc[2][4] = {};

    auto loadFrag = [&](const _Float16* base) -> v16h {
        v8h lo = *(const v8h*)base;
        v8h hi = *(const v8h*)(base + 16);
        v16h f;
#pragma unroll
        for (int e = 0; e < 8; ++e) { f[e] = lo[e]; f[e + 8] = hi[e]; }
        return f;
    };

    auto stageA = [&](int k0, int buf, uint4& av0, uint4& av1) {
        const _Float16* ag = A + (size_t)(rowBlock + ar) * lda + (k0 + ak);
#if HAVE_ASYNC_LDS
        async_cp16(ag,     &As[buf][ar * LDSS + ak]);
        async_cp16(ag + 8, &As[buf][ar * LDSS + ak + 8]);
        (void)av0; (void)av1;
#else
        av0 = *(const uint4*)ag;
        av1 = *(const uint4*)(ag + 8);
        (void)buf;
#endif
    };
    auto storeA = [&](int buf, uint4 av0, uint4 av1) {
#if !HAVE_ASYNC_LDS
        *(uint4*)&As[buf][ar * LDSS + ak]     = av0;
        *(uint4*)&As[buf][ar * LDSS + ak + 8] = av1;
#else
        (void)buf; (void)av0; (void)av1;
#endif
    };
    auto loadB = [&](int k0, uint4& bv0, uint4& bv1) {
        const _Float16* bg = B + (size_t)(k0 + bkp) * ldb + (colBlock + bn);
        bv0 = *(const uint4*)bg;
        bv1 = *(const uint4*)(bg + ldb);
    };
    auto storeB = [&](int buf, uint4 bv0, uint4 bv1) {
        __align__(16) _Float16 t0[8], t1[8];
        *(uint4*)t0 = bv0; *(uint4*)t1 = bv1;
#pragma unroll
        for (int e = 0; e < 8; ++e) {
            union { _Float16 h[2]; unsigned u; } p;
            p.h[0] = t0[e]; p.h[1] = t1[e];
            *(unsigned*)&Bs[buf][(bn + e) * LDSS + bkp] = p.u;
        }
    };
    auto compute = [&](int buf) {
        v16h afrag[2], bfrag[4];
#pragma unroll
        for (int mi = 0; mi < 2; ++mi)
            afrag[mi] = loadFrag(&As[buf][(wm * 32 + mi * 16 + l16) * LDSS + lhalf * 8]);
#pragma unroll
        for (int ni = 0; ni < 4; ++ni)
            bfrag[ni] = loadFrag(&Bs[buf][(wn * 64 + ni * 16 + l16) * LDSS + lhalf * 8]);
#pragma unroll
        for (int mi = 0; mi < 2; ++mi)
#pragma unroll
            for (int ni = 0; ni < 4; ++ni)
                acc[mi][ni] = __builtin_amdgcn_wmma_f32_16x16x32_f16(
                    false, afrag[mi], false, bfrag[ni],
                    (short)0, acc[mi][ni], false, false);
    };

    const int nkt = K / BK;
    uint4 av0, av1, bv0, bv1;

    // prologue: stage tile 0 into buffer 0
    stageA(0, 0, av0, av1);
    storeA(0, av0, av1);
    loadB(0, bv0, bv1);
    storeB(0, bv0, bv1);
    wait_async0();
    __syncthreads();

    for (int kt = 0; kt < nkt; ++kt) {
        const int  cur  = kt & 1;
        const bool more = (kt + 1) < nkt;
        if (more) {
            stageA((kt + 1) * BK, 1 - cur, av0, av1);   // async: lands in other buf
            loadB((kt + 1) * BK, bv0, bv1);
            if (kt + 2 < nkt) {                          // prefetch tile kt+2
                __builtin_prefetch(A + (size_t)(rowBlock + ar) * lda + (kt + 2) * BK + ak, 0, 1);
                __builtin_prefetch(B + (size_t)((kt + 2) * BK + bkp) * ldb + colBlock + bn, 0, 1);
            }
        }
        compute(cur);
        if (more) {
            storeA(1 - cur, av0, av1);
            storeB(1 - cur, bv0, bv1);
            wait_async0();
            __syncthreads();
        }
    }

    // ---- epilogue : C/D layout: vgpr e -> row (lhalf*8+e), col l16 ----
#pragma unroll
    for (int mi = 0; mi < 2; ++mi)
#pragma unroll
        for (int ni = 0; ni < 4; ++ni) {
            const int cg = colBlock + wn * 64 + ni * 16 + l16;
#pragma unroll
            for (int e = 0; e < 8; ++e) {
                const int rg = rowBlock + wm * 32 + mi * 16 + lhalf * 8 + e;
                float v = acc[mi][ni][e];
                if (MODE == 1) {
                    if (rg == cg) v = 0.0f;
                }
                if (MODE == 2) {
                    v = dinv[rg] * (v + 2.0f * xws[(size_t)rg * Ncols + cg]) + bias[cg];
                    if (leaky) v = (v >= 0.0f) ? v : 0.01f * v;
                }
                C[(size_t)rg * ldc + cg] = v;
            }
        }
}

// ---------------------------------------------------------------------------
// small helper kernels
// ---------------------------------------------------------------------------
__global__ void fill_zero_kernel(float* p, size_t count)
{
    size_t i = (size_t)blockIdx.x * blockDim.x + threadIdx.x;
    size_t st = (size_t)gridDim.x * blockDim.x;
    for (; i < count; i += st) p[i] = 0.0f;
}

__global__ void scatter_edges_kernel(const long long* __restrict__ ei,
                                     float* __restrict__ A, int E, int n)
{
    int t = blockIdx.x * blockDim.x + threadIdx.x;
    if (t >= E) return;
    int s = (int)ei[t];        // row 0 = src
    int d = (int)ei[E + t];    // row 1 = dst
    atomicAdd(&A[(size_t)d * n + s], 1.0f);
}

__global__ void deg_dinv_kernel(const float* __restrict__ A, int n,
                                float* __restrict__ dinv)
{
    __shared__ float red[256];
    const int row = blockIdx.x, t = threadIdx.x;
    float s = 0.0f;
    for (int j = t; j < n; j += 256) s += A[(size_t)row * n + j];
    red[t] = s; __syncthreads();
    for (int st = 128; st > 0; st >>= 1) {
        if (t < st) red[t] += red[t + st];
        __syncthreads();
    }
    if (t == 0) {
        float d = red[0] + 2.0f;               // improved: +2I self loops
        dinv[row] = (d > 0.0f) ? rsqrtf(d) : 0.0f;
    }
}

__global__ void tof16_kernel(const float* __restrict__ in,
                             _Float16* __restrict__ out, size_t count)
{
    size_t i = (size_t)blockIdx.x * blockDim.x + threadIdx.x;
    size_t st = (size_t)gridDim.x * blockDim.x;
    for (; i < count; i += st) out[i] = (_Float16)in[i];
}

// A1 = A with diagonal forced to 1 (remove self loops, add unit loops); 2-D grid
__global__ void tof16_unitdiag_kernel(const float* __restrict__ in,
                                      _Float16* __restrict__ out, int n)
{
    int r = blockIdx.y;
    int c = blockIdx.x * blockDim.x + threadIdx.x;
    if (c >= n) return;
    size_t i = (size_t)r * n + c;
    out[i] = (r == c) ? (_Float16)1.0f : (_Float16)in[i];
}

// XWs = dinv[row] * XW ; emit f32 master + f16 operand
__global__ void scale_rows_kernel(const float* __restrict__ xw,
                                  const float* __restrict__ dinv,
                                  float* __restrict__ xws,
                                  _Float16* __restrict__ xws16, int count)
{
    int i = blockIdx.x * blockDim.x + threadIdx.x;
    if (i >= count) return;
    float v = dinv[i >> 7] * xw[i];
    xws[i] = v;
    xws16[i] = (_Float16)v;
}

// score = tanh((x . w) / ||w||), one 128-thread block per node
__global__ void score_kernel(const float* __restrict__ x,
                             const float* __restrict__ w,
                             float* __restrict__ score)
{
    __shared__ float sd[128], sn[128];
    const int row = blockIdx.x, t = threadIdx.x;
    float wv = w[t], xv = x[(size_t)row * 128 + t];
    sd[t] = xv * wv; sn[t] = wv * wv;
    __syncthreads();
    for (int s = 64; s > 0; s >>= 1) {
        if (t < s) { sd[t] += sd[t + s]; sn[t] += sn[t + s]; }
        __syncthreads();
    }
    if (t == 0) score[row] = tanhf(sd[0] * rsqrtf(sn[0]));
}

// full bitonic sort in LDS (n is a power of two, n <= 4096);
// order: score descending, ties -> lower index first (matches lax.top_k)
__global__ __launch_bounds__(1024)
void topk_sort_kernel(const float* __restrict__ score, int n, int k,
                      int* __restrict__ perm, float* __restrict__ gate)
{
    __shared__ float skey[4096];
    __shared__ int   sidx[4096];
    const int tid = threadIdx.x;
    for (int i = tid; i < n; i += 1024) { skey[i] = score[i]; sidx[i] = i; }
    __syncthreads();
    for (int size = 2; size <= n; size <<= 1) {
        for (int stride = size >> 1; stride > 0; stride >>= 1) {
            for (int t = tid; t < (n >> 1); t += 1024) {
                int i = (t << 1) - (t & (stride - 1));
                int j = i + stride;
                bool dir = ((i & size) == 0);
                float ki = skey[i], kj = skey[j];
                int ii = sidx[i], ij = sidx[j];
                bool iBefore = (ki > kj) || (ki == kj && ii < ij);
                if (iBefore != dir) {
                    skey[i] = kj; skey[j] = ki;
                    sidx[i] = ij; sidx[j] = ii;
                }
            }
            __syncthreads();
        }
    }
    for (int r = tid; r < k; r += 1024) { perm[r] = sidx[r]; gate[r] = skey[r]; }
}

// pooled x (f16 operand for the x@W gemm): x16[r,c] = x[perm[r],c] * gate[r]
__global__ void pool_x_kernel(const float* __restrict__ x,
                              const int* __restrict__ perm,
                              const float* __restrict__ gate,
                              _Float16* __restrict__ x16, int k)
{
    int i = blockIdx.x * blockDim.x + threadIdx.x;
    if (i >= k * 128) return;
    int r = i >> 7, c = i & 127;
    x16[i] = (_Float16)(x[(size_t)perm[r] * 128 + c] * gate[r]);
}

// Ap[r,c] = Aaug[perm[r], perm[c]] ; 2-D grid (blockIdx.y = r)
__global__ void pool_A_kernel(const float* __restrict__ Aaug,
                              const int* __restrict__ perm,
                              float* __restrict__ Aout, int k, int np)
{
    int r = blockIdx.y;
    int c = blockIdx.x * blockDim.x + threadIdx.x;
    if (c >= k) return;
    Aout[(size_t)r * k + c] = Aaug[(size_t)perm[r] * np + perm[c]];
}

// xs_j[perm[r], :] += x_up[r, :]   (perm rows are distinct -> race free)
__global__ void unpool_add_kernel(float* __restrict__ xsj,
                                  const int* __restrict__ perm,
                                  const float* __restrict__ xup, int kk)
{
    int i = blockIdx.x * blockDim.x + threadIdx.x;
    if (i >= kk * 128) return;
    int r = i >> 7, c = i & 127;
    xsj[(size_t)perm[r] * 128 + c] += xup[i];
}

// ---------------------------------------------------------------------------
// host-side orchestration
// ---------------------------------------------------------------------------
static inline void* ws_take(char* base, size_t& off, size_t bytes)
{
    off = (off + 255) & ~(size_t)255;
    void* p = base + off;
    off += bytes;
    return p;
}
static inline dim3 g1(size_t count) { return dim3((unsigned)((count + 255) / 256)); }

extern "C" void kernel_launch(void* const* d_in, const int* in_sizes, int n_in,
                              void* d_out, int out_size, void* d_ws, size_t ws_size,
                              hipStream_t stream)
{
    constexpr int N = 4096, E = 65536, H = 128, D = 3;
    const int ns[4] = {4096, 2048, 1024, 512};

    const float*     x_in = (const float*)d_in[0];
    const long long* ei   = (const long long*)d_in[1];   // (2,E) int64
    const float*     Wd   = (const float*)d_in[2];       // (4,128,128)
    const float*     bd   = (const float*)d_in[3];       // (4,128)
    const float*     Wu   = (const float*)d_in[4];       // (3,128,128)
    const float*     bu   = (const float*)d_in[5];       // (3,128)
    const float*     pw   = (const float*)d_in[6];       // (3,128)
    float*           out  = (float*)d_out;

    char* w = (char*)d_ws; size_t off = 0;
    float*     A0    = (float*)ws_take(w, off, (size_t)4096 * 4096 * 4);
    float*     A1m   = (float*)ws_take(w, off, (size_t)2048 * 2048 * 4);
    float*     A2m   = (float*)ws_take(w, off, (size_t)1024 * 1024 * 4);
    float*     A3m   = (float*)ws_take(w, off, (size_t)512 * 512 * 4);
    float*     Aaug  = (float*)ws_take(w, off, (size_t)4096 * 4096 * 4);
    _Float16*  A16   = (_Float16*)ws_take(w, off, (size_t)4096 * 4096 * 2);
    _Float16*  x16   = (_Float16*)ws_take(w, off, (size_t)4096 * 128 * 2);
    _Float16*  w16   = (_Float16*)ws_take(w, off, (size_t)128 * 128 * 2);
    float*     xw32  = (float*)ws_take(w, off, (size_t)4096 * 128 * 4);
    float*     xws32 = (float*)ws_take(w, off, (size_t)4096 * 128 * 4);
    _Float16*  xws16 = (_Float16*)ws_take(w, off, (size_t)4096 * 128 * 2);
    float*     xs0   = (float*)ws_take(w, off, (size_t)4096 * 128 * 4);
    float*     xs1   = (float*)ws_take(w, off, (size_t)2048 * 128 * 4);
    float*     xs2   = (float*)ws_take(w, off, (size_t)1024 * 128 * 4);
    float*     xb0   = (float*)ws_take(w, off, (size_t)2048 * 128 * 4);
    float*     xb1   = (float*)ws_take(w, off, (size_t)2048 * 128 * 4);
    float*     dinv0 = (float*)ws_take(w, off, 4096 * 4);
    float*     dinv1 = (float*)ws_take(w, off, 2048 * 4);
    float*     dinv2 = (float*)ws_take(w, off, 1024 * 4);
    float*     dinv3 = (float*)ws_take(w, off, 512 * 4);
    int*       perm0 = (int*)ws_take(w, off, 2048 * 4);
    int*       perm1 = (int*)ws_take(w, off, 1024 * 4);
    int*       perm2 = (int*)ws_take(w, off, 512 * 4);
    float*     gate  = (float*)ws_take(w, off, 2048 * 4);
    float*     score = (float*)ws_take(w, off, 4096 * 4);
    (void)ws_size; (void)n_in; (void)in_sizes; (void)out_size;

    float* Ams[4]   = {A0, A1m, A2m, A3m};
    float* dinvs[4] = {dinv0, dinv1, dinv2, dinv3};
    int*   perms[3] = {perm0, perm1, perm2};
    float* xss[3]   = {xs0, xs1, xs2};

    // ---------- build dense adjacency ----------
    fill_zero_kernel<<<4096, 256, 0, stream>>>(A0, (size_t)N * N);
    scatter_edges_kernel<<<E / 256, 256, 0, stream>>>(ei, A0, E, N);

    // ---------- down level 0 GCN ----------
    deg_dinv_kernel<<<N, 256, 0, stream>>>(A0, N, dinv0);
    tof16_kernel<<<g1((size_t)N * H), 256, 0, stream>>>(x_in, x16, (size_t)N * H);
    tof16_kernel<<<g1(H * H), 256, 0, stream>>>(Wd, w16, (size_t)H * H);
    gemm_f16<0><<<dim3(1, N / BM), 256, 0, stream>>>(
        x16, H, w16, H, xw32, H, N, H, H, nullptr, nullptr, nullptr, 0);
    scale_rows_kernel<<<g1((size_t)N * H), 256, 0, stream>>>(xw32, dinv0, xws32, xws16, N * H);
    tof16_kernel<<<16384, 256, 0, stream>>>(A0, A16, (size_t)N * N);
    gemm_f16<2><<<dim3(1, N / BM), 256, 0, stream>>>(
        A16, N, xws16, H, xs0, H, N, H, N, dinv0, xws32, bd, 1);
    float* xcur = xs0;

    // ---------- down levels 1..3 : augment -> pool -> GCN ----------
    for (int i = 1; i <= D; ++i) {
        const int np = ns[i - 1], k = ns[i];
        // augment: A1 = A with unit diag; B = A1@A1; zero diag
        tof16_unitdiag_kernel<<<dim3(np / 256, np), 256, 0, stream>>>(Ams[i - 1], A16, np);
        gemm_f16<1><<<dim3(np / BN, np / BM), 256, 0, stream>>>(
            A16, np, A16, np, Aaug, np, np, np, np, nullptr, nullptr, nullptr, 0);
        // top-k pooling
        score_kernel<<<np, 128, 0, stream>>>(xcur, pw + (size_t)(i - 1) * H, score);
        topk_sort_kernel<<<1, 1024, 0, stream>>>(score, np, k, perms[i - 1], gate);
        pool_x_kernel<<<g1((size_t)k * H), 256, 0, stream>>>(xcur, perms[i - 1], gate, x16, k);
        pool_A_kernel<<<dim3(k / 256, k), 256, 0, stream>>>(Aaug, perms[i - 1], Ams[i], k, np);
        // GCN level i
        deg_dinv_kernel<<<k, 256, 0, stream>>>(Ams[i], k, dinvs[i]);
        tof16_kernel<<<g1(H * H), 256, 0, stream>>>(Wd + (size_t)i * H * H, w16, (size_t)H * H);
        gemm_f16<0><<<dim3(1, k / BM), 256, 0, stream>>>(
            x16, H, w16, H, xw32, H, k, H, H, nullptr, nullptr, nullptr, 0);
        scale_rows_kernel<<<g1((size_t)k * H), 256, 0, stream>>>(xw32, dinvs[i], xws32, xws16, k * H);
        tof16_kernel<<<g1((size_t)k * k), 256, 0, stream>>>(Ams[i], A16, (size_t)k * k);
        float* xdst = (i < D) ? xss[i] : xb0;
        gemm_f16<2><<<dim3(1, k / BM), 256, 0, stream>>>(
            A16, k, xws16, H, xdst, H, k, H, k, dinvs[i], xws32, bd + (size_t)i * H, 1);
        xcur = xdst;
    }

    // ---------- up pass ----------
    for (int i = 0; i < D; ++i) {
        const int j = D - 1 - i;
        const int np = ns[j], kk = ns[j + 1];
        unpool_add_kernel<<<g1((size_t)kk * H), 256, 0, stream>>>(xss[j], perms[j], xcur, kk);
        tof16_kernel<<<g1((size_t)np * H), 256, 0, stream>>>(xss[j], x16, (size_t)np * H);
        tof16_kernel<<<g1(H * H), 256, 0, stream>>>(Wu + (size_t)i * H * H, w16, (size_t)H * H);
        gemm_f16<0><<<dim3(1, np / BM), 256, 0, stream>>>(
            x16, H, w16, H, xw32, H, np, H, H, nullptr, nullptr, nullptr, 0);
        scale_rows_kernel<<<g1((size_t)np * H), 256, 0, stream>>>(xw32, dinvs[j], xws32, xws16, np * H);
        tof16_kernel<<<g1((size_t)np * np), 256, 0, stream>>>(Ams[j], A16, (size_t)np * np);
        float* xdst = (i == 0) ? xb1 : ((i == 1) ? xb0 : out);
        gemm_f16<2><<<dim3(1, np / BM), 256, 0, stream>>>(
            A16, np, xws16, H, xdst, H, np, H, np, dinvs[j], xws32, bu + (size_t)i * H,
            (i < D - 1) ? 1 : 0);
        xcur = xdst;
    }
}
